// LTCLayer_87110526697520
// MI455X (gfx1250) — matrix operand order repaired and checked by reading, compiled, not run on hardware
//
#include <hip/hip_runtime.h>
#include <math.h>

typedef __attribute__((ext_vector_type(16))) _Float16 v16h;
typedef __attribute__((ext_vector_type(8)))  _Float16 v8h;
typedef __attribute__((ext_vector_type(8)))  float    v8f;

#define B_DIM 64
#define T_LEN 4096
#define I_DIM 128
#define H_DIM 256

// ---------------------------------------------------------------------------
// Kernel A: x_proj[b,t,h] = x[b,t,:] @ W_in[h,:] + b_in[h]
// M = B*T = 262144 rows, N = 256, K = 128.  HBM roofline: ~400MB -> ~17us.
// Each block owns 8 M-tiles (128 rows). W_in fragments are converted to f16
// registers ONCE per block (kills the 8x redundant L2/cvt traffic), the whole
// 128x128 x slab is staged to LDS f16 in one phase (single barrier), then each
// of 8 waves computes 8 M-tiles x 2 N-tiles of v_wmma_f32_16x16x32_f16.
// ---------------------------------------------------------------------------
#define XA_STR 136        // padded halves per LDS row (272B: 16B aligned)
#define MT_PER_BLK 8

__global__ __launch_bounds__(256)
void ltc_xproj_kernel(const float* __restrict__ x,
                      const float* __restrict__ Win,
                      const float* __restrict__ bin,
                      float* __restrict__ xproj)
{
    __shared__ _Float16 xa[16 * MT_PER_BLK * XA_STR];   // 34816 B
    const int tid  = threadIdx.x;
    const int lane = tid & 31;
    const int wave = tid >> 5;          // 0..7
    const int l15  = lane & 15;
    const int g    = lane >> 4;         // lane half

    const int n0 = wave * 32 + l15;     // first N-tile column for this lane
    const int n1 = n0 + 16;             // second N-tile column

    // --- W_in B-fragments -> registers, once per block --------------------
    // B(32x16 f16): lane = col n, element e -> K = kb + 16g + e
    v16h wf0[4], wf1[4];
    #pragma unroll
    for (int ks = 0; ks < 4; ++ks) {
        const float4* w0 = (const float4*)(Win + (long)n0 * I_DIM + ks * 32 + g * 16);
        const float4* w1 = (const float4*)(Win + (long)n1 * I_DIM + ks * 32 + g * 16);
        #pragma unroll
        for (int q = 0; q < 4; ++q) {
            float4 a = w0[q], b = w1[q];
            wf0[ks][4*q+0] = (_Float16)a.x; wf0[ks][4*q+1] = (_Float16)a.y;
            wf0[ks][4*q+2] = (_Float16)a.z; wf0[ks][4*q+3] = (_Float16)a.w;
            wf1[ks][4*q+0] = (_Float16)b.x; wf1[ks][4*q+1] = (_Float16)b.y;
            wf1[ks][4*q+2] = (_Float16)b.z; wf1[ks][4*q+3] = (_Float16)b.w;
        }
    }

    // --- Stage 128 x-rows f32 -> f16 LDS (coalesced, one barrier) ---------
    const float* xrow = x + (long)blockIdx.x * (16 * MT_PER_BLK) * I_DIM;
    #pragma unroll 4
    for (int i = tid; i < 16 * MT_PER_BLK * I_DIM; i += 256) {
        int m = i >> 7, k = i & 127;
        xa[m * XA_STR + k] = (_Float16)xrow[m * I_DIM + k];
    }
    __syncthreads();

    const float bi0 = bin[n0];
    const float bi1 = bin[n1];

    // --- 8 M-tiles, no further barriers -----------------------------------
    #pragma unroll
    for (int mt = 0; mt < MT_PER_BLK; ++mt) {
        v8f acc0 = {};
        v8f acc1 = {};
        const int mrow = mt * 16 + l15;
        #pragma unroll
        for (int ks = 0; ks < 4; ++ks) {
            const int kb = ks * 32;
            // A(16x32 f16): row = l15; element e -> K=kb+16g+e
            v8h alo = *(const v8h*)&xa[mrow * XA_STR + kb + g * 16];
            v8h ahi = *(const v8h*)&xa[mrow * XA_STR + kb + g * 16 + 8];
            v16h a;
            #pragma unroll
            for (int e = 0; e < 8; ++e) { a[e] = alo[e]; a[e + 8] = ahi[e]; }
            acc0 = __builtin_amdgcn_wmma_f32_16x16x32_f16(false, a, false, wf0[ks],
                                                          (short)0, acc0, false, false);
            acc1 = __builtin_amdgcn_wmma_f32_16x16x32_f16(false, a, false, wf1[ks],
                                                          (short)0, acc1, false, false);
        }
        const long rowBase = ((long)blockIdx.x * MT_PER_BLK + mt) * 16;
        #pragma unroll
        for (int r = 0; r < 8; ++r) {
            const int  m   = r + 8 * g;            // C/D layout: VGPR r -> row r+8g
            const long off = (rowBase + m) * (long)H_DIM;
            xproj[off + n0] = acc0[r] + bi0;
            xproj[off + n1] = acc1[r] + bi1;
        }
    }
}

// ---------------------------------------------------------------------------
// Kernel B: recurrent scan. Batch rows independent -> 4 WGs x 16 rows each,
// 16 waves/WG (one 16x16 N-tile each). W_rec fragments register-resident;
// h kept f32 in VGPRs + f16 ping-pong mirror in LDS (ONE barrier per layer).
// ---------------------------------------------------------------------------
#define HSTR 264   // padded halves per h row: 528B -> stride == 4 dwords mod 64 banks

__global__ __launch_bounds__(512)
void ltc_scan_kernel(const float* __restrict__ xproj,
                     const float* __restrict__ Wrec,
                     const float* __restrict__ brec,
                     const float* __restrict__ tau,
                     float* __restrict__ hout)
{
    __shared__ _Float16 hl[2][16 * HSTR];           // ping-pong, 16896 B
    const int tid   = threadIdx.x;
    const int lane  = tid & 31;
    const int wave  = tid >> 5;         // 0..15 -> N tile
    const int l15   = lane & 15;
    const int g     = lane >> 4;
    const int col   = wave * 16 + l15;  // output neuron owned by this lane
    const long bBase = (long)blockIdx.x * 16;

    // Per-column constants
    const float br = brec[col];
    float tc = tau[col];
    tc = fminf(fmaxf(tc, 0.1f), 5.0f);
    const float sc = 0.1f / tc;

    // W_rec B-fragments -> registers once: B[k][n] = Wrec[n][k]
    v16h wf[8];
    #pragma unroll
    for (int ks = 0; ks < 8; ++ks) {
        const float4* w4 = (const float4*)(Wrec + (long)col * H_DIM + ks * 32 + g * 16);
        #pragma unroll
        for (int q = 0; q < 4; ++q) {
            float4 v = w4[q];
            wf[ks][4*q+0] = (_Float16)v.x; wf[ks][4*q+1] = (_Float16)v.y;
            wf[ks][4*q+2] = (_Float16)v.z; wf[ks][4*q+3] = (_Float16)v.w;
        }
    }

    // h0 = 0 (only buffer 0 is read first)
    for (int i = tid; i < 16 * HSTR; i += 512) hl[0][i] = (_Float16)0.0f;
    v8f h = {};
    __syncthreads();

    const long TH = (long)T_LEN * H_DIM;
    const float* xp = xproj + col;
    int cur = 0;

    for (int t = 0; t < T_LEN; ++t) {
        // Input projection tile for this step (shared by both inner layers)
        float xt[8];
        #pragma unroll
        for (int r = 0; r < 8; ++r) {
            const int m = r + 8 * g;
            xt[r] = xp[(bBase + m) * TH + (long)t * H_DIM];
        }
        if (t + 1 < T_LEN) {   // prefetch next step -> global_prefetch_b8
            #pragma unroll
            for (int r = 0; r < 8; ++r) {
                const int m = r + 8 * g;
                __builtin_prefetch(&xp[(bBase + m) * TH + (long)(t + 1) * H_DIM], 0, 0);
            }
        }

        #pragma unroll
        for (int layer = 0; layer < 2; ++layer) {
            const _Float16* src = hl[cur];
            _Float16*       dst = hl[cur ^ 1];

            // acc = h @ W_rec^T for this wave's 16x16 tile (K = 256)
            v8f acc = {};
            #pragma unroll
            for (int ks = 0; ks < 8; ++ks) {
                const int kb = ks * 32;
                v8h alo = *(const v8h*)&src[l15 * HSTR + kb + g * 16];
                v8h ahi = *(const v8h*)&src[l15 * HSTR + kb + g * 16 + 8];
                v16h a;
                #pragma unroll
                for (int e = 0; e < 8; ++e) { a[e] = alo[e]; a[e + 8] = ahi[e]; }
                acc = __builtin_amdgcn_wmma_f32_16x16x32_f16(false, a, false, wf[ks],
                                                             (short)0, acc, false, false);
            }
            // Leaky LTC update (f32, registers)
            #pragma unroll
            for (int r = 0; r < 8; ++r) {
                const float pre = xt[r] + acc[r] + br;
                const float act = tanhf(pre);
                h[r] = h[r] + sc * (act - h[r]);
            }
            // Publish f16 h into the OTHER buffer; reads of `src` all precede
            // this barrier, writes to `src` resume only next iteration.
            #pragma unroll
            for (int r = 0; r < 8; ++r)
                dst[(r + 8 * g) * HSTR + col] = (_Float16)h[r];
            __syncthreads();
            cur ^= 1;
        }
    }

    // Final hidden state -> d_out [B, H] f32
    #pragma unroll
    for (int r = 0; r < 8; ++r) {
        const int m = r + 8 * g;
        hout[(bBase + m) * H_DIM + col] = h[r];
    }
}

// ---------------------------------------------------------------------------
extern "C" void kernel_launch(void* const* d_in, const int* in_sizes, int n_in,
                              void* d_out, int out_size, void* d_ws, size_t ws_size,
                              hipStream_t stream)
{
    const float* x    = (const float*)d_in[0];   // [64,4096,128]
    const float* Win  = (const float*)d_in[1];   // [256,128]
    const float* bin  = (const float*)d_in[2];   // [256]
    const float* Wrec = (const float*)d_in[3];   // [256,256]
    const float* brec = (const float*)d_in[4];   // [256]
    const float* tau  = (const float*)d_in[5];   // [256]
    float* xproj = (float*)d_ws;                 // [64*4096, 256] f32 scratch
    float* out   = (float*)d_out;                // [64, 256] f32

    (void)in_sizes; (void)n_in; (void)out_size; (void)ws_size;

    // Input projection GEMM: 2048 blocks x 8 M-tiles
    ltc_xproj_kernel<<<(B_DIM * T_LEN) / (16 * MT_PER_BLK), 256, 0, stream>>>(x, Win, bin, xproj);
    // Recurrent scan: 4 independent batch slices
    ltc_scan_kernel<<<B_DIM / 16, 512, 0, stream>>>(xproj, Wrec, brec, tau, out);
}
